// NLSApproxSpace_58050777973396
// MI455X (gfx1250) — matrix operand order repaired and checked by reading, not compiled, not run on hardware
//
#include <hip/hip_runtime.h>
#include <hip/hip_bf16.h>

// ---------------------------------------------------------------------------
// MI455X (gfx1250) non-local search.
// Shapes: B=1,T=3,C=64,H=W=192, NHEADS=1, K=7, PS=7, WS=8, WT=1,
// STRIDE0_A=8, STRIDE0=4.
// Structure (derived from the reference):
//  * t0 = clip(qt-1,0,T-st) = 0 always -> ct = {0,1,2} for every query
//  * offs = arange(8)-4, stride 8      -> candidate 8x8 windows tile H,W
//  * candidates depend only on (i,j)   -> qt=0,1,2 share one candidate set
// => exact search per coarse cell (i,j) is a GEMM:
//      D[192 cand x 3 query] = A[192 x 3136] * B[3136 x 3(pad 16)]
//    -> v_wmma_f32_16x16x32_bf16, 12 M-tiles x 98 K-steps per cell.
// Staging uses CDNA5 async direct-to-LDS loads (ASYNCcnt-tracked), the
// WMMA loop uses only immediate-offset ds_load_b128 off invariant bases.
// Refine: 48384 independent 3136-length gathers -> VALU bf16 FMA.
// ---------------------------------------------------------------------------

typedef __bf16 bf16_t;
typedef __attribute__((ext_vector_type(16))) __bf16 v16bf;
typedef __attribute__((ext_vector_type(8)))  float  v8f;

union Frag32 {            // 16 bf16 = 32B = two 16B LDS loads
  v16bf v;
  uint4 q[2];
};

__device__ __forceinline__ int refl(int i, int L) {
  i = i < 0 ? -i : i;
  return (i >= L) ? (2 * (L - 1) - i) : i;
}

// CDNA5 async global->LDS copy of one 16B chunk per lane (GVS addressing:
// mem = SADDR_u64 + VADDR_u32 + ioffset). VDST holds the LDS byte address.
// Tracked by ASYNCcnt; pair with s_wait_asynccnt + barrier before use.
__device__ __forceinline__ void async_copy_b128(unsigned lds_addr,
                                                unsigned goffset,
                                                unsigned long long gbase) {
  asm volatile("global_load_async_to_lds_b128 %0, %1, %2"
               :: "v"(lds_addr), "v"(goffset), "s"(gbase)
               : "memory");
}
__device__ __forceinline__ void wait_async0() {
  asm volatile("s_wait_asynccnt 0x0" ::: "memory");
}

#define T_ 3
#define C_ 64
#define H_ 192
#define W_ 192
#define NHC 24              // coarse grid 192/8
#define NHF 48              // fine grid   192/4
#define QF  (T_*NHF*NHF)    // 6912 fine queries
#define KTOP 7
#define VOL_BF16 (T_*H_*W_*C_)                 // 7077888 elements
#define V1T_OFF  ((size_t)VOL_BF16 * 2)        // bytes
#define CIND_OFF ((size_t)VOL_BF16 * 4)        // bytes

// ---------------------------------------------------------------------------
// Kernel 1: (B,T,C,H,W) f32 -> (T,H,W,C) bf16 via LDS tile transpose.
// ---------------------------------------------------------------------------
__global__ __launch_bounds__(256) void convert_kernel(
    const float* __restrict__ v0, const float* __restrict__ v1,
    bf16_t* __restrict__ v0T, bf16_t* __restrict__ v1T) {
  __shared__ float tile[64][33];
  int blk = blockIdx.x;
  int wchunk = blk % 6;
  int h = (blk / 6) % H_;
  int t = (blk / (6 * H_)) % T_;
  int arr = blk / (6 * H_ * T_);
  const float* src = arr ? v1 : v0;
  bf16_t* dst = arr ? v1T : v0T;
  int w0 = wchunk * 32;
  int tid = threadIdx.x;
#pragma unroll
  for (int it = 0; it < 8; ++it) {           // coalesced reads along w
    int idx = it * 256 + tid;
    int c = idx >> 5, wl = idx & 31;
    tile[c][wl] = src[(((size_t)t * C_ + c) * H_ + h) * W_ + w0 + wl];
  }
  __syncthreads();
#pragma unroll
  for (int it = 0; it < 8; ++it) {           // coalesced writes along c
    int idx = it * 256 + tid;
    int wl = idx >> 6, c = idx & 63;
    dst[(((size_t)t * H_ + h) * W_ + w0 + wl) * C_ + c] = (bf16_t)tile[c][wl];
  }
}

// ---------------------------------------------------------------------------
// Kernel 2: exact search via async-to-LDS staging + WMMA + in-LDS top-7.
// One block per coarse cell (576 blocks), 192 threads = 6 waves.
// Wave w owns candidate M-tiles {2w, 2w+1} (16 candidates each, 192 total).
// LDS: Aq[3][49][64] (18.8KB) + Reg[3][14][14][32ch] (37.6KB) +
//      Dists[3][192] (2.3KB) = 58.8KB static (< 64KB).
// ---------------------------------------------------------------------------
__global__ __launch_bounds__(192) void exact_search_kernel(
    const bf16_t* __restrict__ v0T, const bf16_t* __restrict__ v1T,
    int* __restrict__ cinds) {
  __shared__ __align__(16) bf16_t Aq[T_][49][C_];
  __shared__ __align__(16) bf16_t Reg[T_][14][14][32];
  __shared__ float Dists[T_][192];

  const int cell = blockIdx.x;             // 0..575
  const int ci = cell / NHC, cj = cell % NHC;
  const int qh = ci * 8, qw = cj * 8;
  const int tid = threadIdx.x;

  // Generic LDS pointers carry the LDS byte offset in their low 32 bits.
  const unsigned aq_lds  = (unsigned)(uintptr_t)&Aq[0][0][0];
  const unsigned reg_lds = (unsigned)(uintptr_t)&Reg[0][0][0][0];
  const unsigned long long v0g = (unsigned long long)(uintptr_t)v0T;
  const unsigned long long v1g = (unsigned long long)(uintptr_t)v1T;

  // Stage query patches asynchronously: 147 (qt,p) pixels * 8 x 16B chunks.
  for (int e = tid; e < T_ * 49 * 8; e += 192) {
    int u = e & 7, t2 = e >> 3;
    int p = t2 % 49, qt = t2 / 49;
    int y = refl(qh + (p / 7 - 3), H_);
    int x = refl(qw + (p % 7 - 3), W_);
    unsigned lds = aq_lds + (unsigned)((t2 * C_ + u * 8) * 2);
    unsigned gof = (unsigned)(((((unsigned)qt * H_ + y) * W_ + x) * C_ + u * 8) * 2);
    async_copy_b128(lds, gof, v0g);
  }

  const int wave = tid >> 5;                // 0..5
  const int lane = tid & 31;
  const int lm = lane & 15;                 // M-row (A) / N-col (B) slot
  const int half = lane >> 4;               // K-half select per ISA layout
  const int nq = lm < 3 ? lm : 2;           // clamp query column (N pad)

  // Per-lane candidate coords (fixed): m -> ts=m/64, wi=(m/8)&7, wj=m&7.
  // Region row index for patch offset dr'=0..6 is simply wi + dr'.
  int m0 = 32 * wave + lm, m1 = m0 + 16;
  int ts0 = m0 >> 6, wi0 = (m0 >> 3) & 7, wj0 = m0 & 7;
  int ts1 = m1 >> 6, wi1 = (m1 >> 3) & 7, wj1 = m1 & 7;

  // Loop-invariant LDS fragment bases (immediate offsets do the rest).
  const bf16_t* abase0 = &Reg[ts0][wi0][wj0][8 * half];
  const bf16_t* abase1 = &Reg[ts1][wi1][wj1][8 * half];
  const bf16_t* bbase  = &Aq[nq][0][8 * half];

  v8f acc0 = {0.f, 0.f, 0.f, 0.f, 0.f, 0.f, 0.f, 0.f};
  v8f acc1 = {0.f, 0.f, 0.f, 0.f, 0.f, 0.f, 0.f, 0.f};

  for (int chalf = 0; chalf < 2; ++chalf) {
    __syncthreads();   // previous Reg readers done before async writes land
    const int c0 = chalf * 32;
    // Stage candidate region slice: 588 pixels * 4 x 16B chunks, async.
    for (int e = tid; e < T_ * 14 * 14 * 4; e += 192) {
      int u = e & 3, px = e >> 2;
      int cc = px % 14, rr = (px / 14) % 14, ts = px / 196;
      int y = refl(qh + rr - 7, H_);
      int x = refl(qw + cc - 7, W_);
      unsigned lds = reg_lds + (unsigned)((px * 32 + u * 8) * 2);
      unsigned gof =
          (unsigned)(((((unsigned)ts * H_ + y) * W_ + x) * C_ + c0 + u * 8) * 2);
      async_copy_b128(lds, gof, v1g);
    }
    wait_async0();     // also covers the Aq asyncs on the first pass
    __syncthreads();

    const bf16_t* bp = bbase + c0;
#pragma unroll
    for (int dr = 0; dr < 7; ++dr) {
#pragma unroll
      for (int dc = 0; dc < 7; ++dc) {
        const int aoff = dr * (14 * 32) + dc * 32;   // elements
        const int boff = (dr * 7 + dc) * C_;         // elements
        Frag32 fb, fa0, fa1;
        fb.q[0]  = *(const uint4*)(bp + boff);
        fb.q[1]  = *(const uint4*)(bp + boff + 16);
        fa0.q[0] = *(const uint4*)(abase0 + aoff);
        fa0.q[1] = *(const uint4*)(abase0 + aoff + 16);
        fa1.q[0] = *(const uint4*)(abase1 + aoff);
        fa1.q[1] = *(const uint4*)(abase1 + aoff + 16);
        // D = A*B + C  (neg_a, A, neg_b, B, c_mod, C, reuse_a, reuse_b)
        acc0 = __builtin_amdgcn_wmma_f32_16x16x32_bf16(
            false, fa0.v, false, fb.v, (short)0, acc0, false, false);
        acc1 = __builtin_amdgcn_wmma_f32_16x16x32_bf16(
            false, fa1.v, false, fb.v, (short)0, acc1, false, false);
      }
    }
  }
  __syncthreads();

  // D layout: lane n = lm (query col), VGPR r -> row (r + 8*half) in tile.
  if (lm < 3) {
#pragma unroll
    for (int r = 0; r < 8; ++r) {
      Dists[lm][32 * wave + 8 * half + r] = acc0[r];
      Dists[lm][32 * wave + 16 + 8 * half + r] = acc1[r];
    }
  }
  __syncthreads();

  // Top-7 per qt (stable: strict '>' keeps earliest s on ties, like top_k).
  if (tid < 3) {
    int qt = tid;
    float bv[KTOP];
    int bs[KTOP];
#pragma unroll
    for (int k = 0; k < KTOP; ++k) { bv[k] = -3.4e38f; bs[k] = 0; }
    for (int s = 0; s < 192; ++s) {
      float v = Dists[qt][s];
      if (v > bv[KTOP - 1]) {
        int k = KTOP - 1;
        while (k > 0 && v > bv[k - 1]) { bv[k] = bv[k - 1]; bs[k] = bs[k - 1]; --k; }
        bv[k] = v; bs[k] = s;
      }
    }
    int* out = cinds + (size_t)(((qt * NHC + ci) * NHC + cj) * KTOP) * 3;
#pragma unroll
    for (int k = 0; k < KTOP; ++k) {
      int s = bs[k];
      int ts = s >> 6, wi = (s >> 3) & 7, wj = s & 7;
      out[3 * k + 0] = ts;                 // ct (raw, no reflect — ref keeps raw)
      out[3 * k + 1] = qh + wi - 4;        // ch
      out[3 * k + 2] = qw + wj - 4;        // cw
    }
  }
}

// ---------------------------------------------------------------------------
// Kernel 3: upsample indices + refine dists. One wave per (fine q, k):
// 48384 waves, lanes split the 64 channels; gather-bound -> VALU FMA.
// d_out = [dists_f (6912*7 f32)] ++ [inds_f (6912*7*3) as f32 values].
// ---------------------------------------------------------------------------
__global__ __launch_bounds__(128) void refine_kernel(
    const bf16_t* __restrict__ v0T, const bf16_t* __restrict__ v1T,
    const int* __restrict__ cinds, float* __restrict__ out) {
  int gw = blockIdx.x * 4 + (threadIdx.x >> 5);   // 0..48383
  int lane = threadIdx.x & 31;
  int q = gw / KTOP, k = gw % KTOP;
  int ft = q / (NHF * NHF);
  int rem = q % (NHF * NHF);
  int fi = rem / NHF, fj = rem % NHF;
  int fh = fi * 4, fw = fj * 4;

  int it, ih, iw;
  if (k == 0) {                                   // candidate 0 = self
    it = ft; ih = fh; iw = fw;
  } else {
    int ci2 = fi >> 1, cj2 = fj >> 1;             // SCALE=2; 47/2=23 <= nH-1
    const int* b =
        cinds + (size_t)((((ft * NHC + ci2) * NHC + cj2) * KTOP + k)) * 3;
    it = b[0];
    ih = refl(b[1] + (fh - ci2 * 8), H_);
    iw = refl(b[2] + (fw - cj2 * 8), W_);
  }

  float sum = 0.f;
#pragma unroll
  for (int dr = -3; dr <= 3; ++dr) {
    int y0 = refl(fh + dr, H_);
    int y1 = refl(ih + dr, H_);
    const bf16_t* arow = v0T + (((size_t)ft * H_ + y0) * W_) * C_;
    const bf16_t* brow = v1T + (((size_t)it * H_ + y1) * W_) * C_;
    __builtin_prefetch(brow + (size_t)refl(iw, W_) * C_, 0, 0);  // gather target
#pragma unroll
    for (int dc = -3; dc <= 3; ++dc) {
      const bf16_t* a = arow + (size_t)refl(fw + dc, W_) * C_;
      const bf16_t* b = brow + (size_t)refl(iw + dc, W_) * C_;
      sum += (float)a[lane] * (float)b[lane] +
             (float)a[lane + 32] * (float)b[lane + 32];
    }
  }
#pragma unroll
  for (int off = 16; off > 0; off >>= 1) sum += __shfl_xor(sum, off, 32);

  if (lane == 0) {
    out[gw] = sum;                                     // dists_f
    float* oi = out + (size_t)QF * KTOP + (size_t)gw * 3;
    oi[0] = (float)it; oi[1] = (float)ih; oi[2] = (float)iw;  // inds_f
  }
}

// ---------------------------------------------------------------------------
extern "C" void kernel_launch(void* const* d_in, const int* in_sizes, int n_in,
                              void* d_out, int out_size, void* d_ws, size_t ws_size,
                              hipStream_t stream) {
  (void)in_sizes; (void)n_in; (void)out_size; (void)ws_size;
  const float* vid0 = (const float*)d_in[0];
  const float* vid1 = (const float*)d_in[1];
  // d_in[2] = flows (all zeros, unused by the reference result path)

  char* ws = (char*)d_ws;
  bf16_t* v0T = (bf16_t*)ws;                       // 14,155,776 B
  bf16_t* v1T = (bf16_t*)(ws + V1T_OFF);           // 14,155,776 B
  int* cinds = (int*)(ws + CIND_OFF);              //    145,152 B
  float* out = (float*)d_out;

  convert_kernel<<<2 * T_ * H_ * 6, 256, 0, stream>>>(vid0, vid1, v0T, v1T);
  exact_search_kernel<<<NHC * NHC, 192, 0, stream>>>(v0T, v1T, cinds);
  refine_kernel<<<(QF * KTOP) / 4, 128, 0, stream>>>(v0T, v1T, cinds, out);
}
